// SimpleFC_65317862637954
// MI455X (gfx1250) — matrix operand order repaired and checked by reading, not compiled, not run on hardware
//
#include <hip/hip_runtime.h>

typedef __attribute__((ext_vector_type(16))) _Float16 v16h;
typedef __attribute__((ext_vector_type(8)))  float    v8f;

#define B_   512
#define T_   200
#define NI_  784
#define NH_  100
#define NO_  10
#define NKP  800          // K padded to multiple of 32
#define NHP  112          // NH padded to multiple of 16
#define NT   7            // N tiles (112/16)
#define ROWS (B_ * T_)    // 102400 rows, 6400 M-tiles

// ---------------------------------------------------------------------------
// Kernel 0: convert W1 (f32 [100][784]) -> zero-padded f16 [112][800]
// ---------------------------------------------------------------------------
__global__ __launch_bounds__(256)
void snn_convert_w1(const float* __restrict__ W1, _Float16* __restrict__ W1h) {
    int i = blockIdx.x * 256 + threadIdx.x;
    if (i >= NHP * NKP) return;
    int n = i / NKP;
    int k = i - n * NKP;
    W1h[i] = (n < NH_ && k < NI_) ? (_Float16)W1[n * NI_ + k] : (_Float16)0.0f;
}

// ---------------------------------------------------------------------------
// Kernel 1: CUR1[t][b][h] = x[b][t][:] @ W1^T + b1   via v_wmma_f32_16x16x32_f16
// 1 wave = one 16-row M tile x all 7 N tiles. 4 waves / block.
// ---------------------------------------------------------------------------
__global__ __launch_bounds__(128)
void snn_gemm1(const float* __restrict__ x, const _Float16* __restrict__ W1h,
               const float* __restrict__ b1, float* __restrict__ cur1) {
    const int lane  = threadIdx.x & 31;
    const int wave  = threadIdx.x >> 5;
    const int tileM = blockIdx.x * 4 + wave;        // 0..6399
    const int m     = lane & 15;
    const int grp   = lane >> 4;                    // 0 or 1
    const long row0 = (long)tileM * 16;
    const float* xr = x + (row0 + m) * (long)NI_;   // this lane's A row

    v8f acc[NT];
    #pragma unroll
    for (int n = 0; n < NT; ++n) acc[n] = (v8f){0,0,0,0,0,0,0,0};

    // main K loop: 24 full 32-wide steps
    for (int k0 = 0; k0 < 768; k0 += 32) {
        v16h a;
        #pragma unroll
        for (int j = 0; j < 8; ++j) {               // j = A VGPR index (pair of halves)
            const int K = grp * 16 + 2 * j;
            const float2 f = *(const float2*)(xr + k0 + K);
            a[2 * j]     = (_Float16)f.x;
            a[2 * j + 1] = (_Float16)f.y;
        }
        #pragma unroll
        for (int n = 0; n < NT; ++n) {
            const int c = n * 16 + m;               // B column == W1 row
            const v16h bf = *(const v16h*)(W1h + (long)c * NKP + k0 + grp * 16);
            acc[n] = __builtin_amdgcn_wmma_f32_16x16x32_f16(
                false, a, false, bf, (short)0, acc[n], false, false);
        }
    }
    // K tail: k0 = 768, only K < 16 valid (784 = 768+16); only grp 0 holds valid K
    {
        const int k0 = 768;
        v16h a;
        #pragma unroll
        for (int j = 0; j < 8; ++j) {
            if (grp == 0) {
                const int K = 2 * j;                // <= 15, pair fully in-bounds
                const float2 f = *(const float2*)(xr + k0 + K);
                a[2 * j]     = (_Float16)f.x;
                a[2 * j + 1] = (_Float16)f.y;
            } else {
                a[2 * j]     = (_Float16)0.0f;
                a[2 * j + 1] = (_Float16)0.0f;
            }
        }
        #pragma unroll
        for (int n = 0; n < NT; ++n) {
            const int c = n * 16 + m;
            const v16h bf = *(const v16h*)(W1h + (long)c * NKP + k0 + grp * 16);
            acc[n] = __builtin_amdgcn_wmma_f32_16x16x32_f16(
                false, a, false, bf, (short)0, acc[n], false, false);
        }
    }

    // store: D lane holds col = lane&15, VGPR v holds row = v + 8*grp
    #pragma unroll
    for (int n = 0; n < NT; ++n) {
        const int c = n * 16 + m;
        if (c < NH_) {
            const float bias = b1[c];
            #pragma unroll
            for (int v = 0; v < 8; ++v) {
                const long rr = row0 + v + 8 * grp; // global row = b*T + t
                const long b  = rr / T_;
                const int  t  = (int)(rr - b * T_);
                cur1[((long)t * B_ + b) * NH_ + c] = acc[n][v] + bias;
            }
        }
    }
}

// ---------------------------------------------------------------------------
// Kernel 2: sequential LIF scan. One wave per batch element (recurrence is
// independent across b). Lanes 0..24 own 4 hidden channels each; layer-2
// output via butterfly shfl_xor reduction; lanes 0..9 own the 10 outputs.
// ---------------------------------------------------------------------------
__global__ __launch_bounds__(256)
void snn_scan(const float* __restrict__ cur1, const float* __restrict__ W2,
              const float* __restrict__ b2, const float* __restrict__ beta_p,
              float* __restrict__ out) {
    const int lane = threadIdx.x & 31;
    const int wave = threadIdx.x >> 5;
    const int b    = blockIdx.x * 8 + wave;         // 0..511
    const float beta = *beta_p;
    const int  h0  = lane * 4;
    const bool act = (lane < 25);                   // 25 lanes * 4 = 100 channels

    float w2r[NO_][4];
    #pragma unroll
    for (int o = 0; o < NO_; ++o)
        #pragma unroll
        for (int j = 0; j < 4; ++j)
            w2r[o][j] = act ? W2[o * NH_ + h0 + j] : 0.0f;

    const float bias2 = (lane < NO_) ? b2[lane] : 0.0f;
    float mem1[4] = {0.f, 0.f, 0.f, 0.f};
    float mem2 = 0.0f;

    for (int t = 0; t < T_; ++t) {
        const float* cb = cur1 + ((long)t * B_ + b) * NH_;
        float cur[4] = {0.f, 0.f, 0.f, 0.f};
        if (act) {
            const float4 f = *(const float4*)(cb + h0);
            cur[0] = f.x; cur[1] = f.y; cur[2] = f.z; cur[3] = f.w;
        }
        float spk[4];
        #pragma unroll
        for (int j = 0; j < 4; ++j) {
            const float rst = (mem1[j] > 1.0f) ? 1.0f : 0.0f;   // reset from mem_prev
            mem1[j] = beta * mem1[j] + cur[j] - rst;
            spk[j]  = (mem1[j] > 1.0f) ? 1.0f : 0.0f;
        }
        float part[NO_];
        #pragma unroll
        for (int o = 0; o < NO_; ++o)
            part[o] = spk[0] * w2r[o][0] + spk[1] * w2r[o][1]
                    + spk[2] * w2r[o][2] + spk[3] * w2r[o][3];
        // wave32 butterfly reduction (all lanes execute)
        #pragma unroll
        for (int o = 0; o < NO_; ++o) {
            #pragma unroll
            for (int off = 16; off > 0; off >>= 1)
                part[o] += __shfl_xor(part[o], off, 32);
        }
        if (lane < NO_) {
            const float cur2 = part[lane] + bias2;
            const float rst2 = (mem2 > 1.0f) ? 1.0f : 0.0f;
            mem2 = beta * mem2 + cur2 - rst2;
            const float spk2 = (mem2 > 1.0f) ? 1.0f : 0.0f;
            const long idx = ((long)t * B_ + b) * NO_ + lane;
            out[idx] = spk2;                                   // spk2_rec
            out[(long)T_ * B_ * NO_ + idx] = mem2;             // mem2_rec
        }
    }
}

// ---------------------------------------------------------------------------
extern "C" void kernel_launch(void* const* d_in, const int* in_sizes, int n_in,
                              void* d_out, int out_size, void* d_ws, size_t ws_size,
                              hipStream_t stream) {
    const float* x    = (const float*)d_in[0];   // [512,200,784]
    const float* W1   = (const float*)d_in[1];   // [100,784]
    const float* b1   = (const float*)d_in[2];   // [100]
    const float* W2   = (const float*)d_in[3];   // [10,100]
    const float* b2   = (const float*)d_in[4];   // [10]
    const float* beta = (const float*)d_in[5];   // scalar
    float* out = (float*)d_out;                  // [2,200,512,10]

    _Float16* W1h = (_Float16*)d_ws;                           // 112*800*2 = 179200 B
    float* cur1   = (float*)((char*)d_ws + (size_t)NHP * NKP * 2); // 102400*100*4 = 41 MB

    snn_convert_w1<<<(NHP * NKP + 255) / 256, 256, 0, stream>>>(W1, W1h);
    snn_gemm1<<<(ROWS / 16) / 4, 128, 0, stream>>>(x, W1h, b1, cur1);
    snn_scan<<<B_ / 8, 256, 0, stream>>>(cur1, W2, b2, beta, out);
}
